// VisionAttention_63891933495385
// MI455X (gfx1250) — matrix operand order repaired and checked by reading, not compiled, not run on hardware
//
#include <hip/hip_runtime.h>

// ---------------------------------------------------------------------------
// CDNA5 (gfx1250) wave32 WMMA implementation of the vision-attention block:
//   qkv = hs @ w_qkv + b ; RoPE(q,k) ; per-(group,head) softmax attention ;
//   out = attn @ w_proj + b_proj
// All GEMM-shaped math goes through v_wmma_f32_16x16x32_bf16; tile staging
// uses async global->LDS (ASYNCcnt) when the toolchain exposes the builtin.
// ---------------------------------------------------------------------------

typedef __attribute__((ext_vector_type(16))) __bf16 v16bf;
typedef __attribute__((ext_vector_type(8)))  __bf16 v8bf;
typedef __attribute__((ext_vector_type(4)))  __bf16 v4bf;
typedef __attribute__((ext_vector_type(2)))  __bf16 v2bf;
typedef __attribute__((ext_vector_type(8)))  float  v8f;

#define S_TOK   16384
#define DIM     1280
#define NH      16
#define HD      80      // head dim
#define NGRP    16
#define LSEQ    1024
#define NQKV    3840

#if defined(__HIP_DEVICE_COMPILE__) && \
    __has_builtin(__builtin_amdgcn_global_load_async_to_lds_b128)
#define ASYNC_LDS 1
#else
#define ASYNC_LDS 0
#endif

#if ASYNC_LDS
// Builtin parameter type (from hipcc diagnostic): pointer to int vector_size(16),
// in __device__ (AS1) / __shared__ (AS3) address spaces.
typedef __attribute__((__vector_size__(4 * sizeof(int)))) int a_v4i;
typedef __attribute__((address_space(1))) a_v4i* gp_v4i;
typedef __attribute__((address_space(3))) a_v4i* lp_v4i;
__device__ __forceinline__ void async_cp16(const __bf16* g, __bf16* l) {
  __builtin_amdgcn_global_load_async_to_lds_b128(
      (gp_v4i)(void*)g, (lp_v4i)(void*)l, 0, 0);
}
__device__ __forceinline__ void async_wait0() {
  asm volatile("s_wait_asynccnt 0x0" ::: "memory");
}
#endif

__device__ __forceinline__ v8f wmma_bf16(v16bf a, v16bf b, v8f c) {
  return __builtin_amdgcn_wmma_f32_16x16x32_bf16(
      /*neg_a=*/false, a, /*neg_b=*/false, b,
      /*c_mod=*/(short)0, c, /*reuse_a=*/false, /*reuse_b=*/false);
}

// A-fragment (16x32 bf16, M x K): lane holds row m = lane&15.
// The 16 halves split into two contiguous 16-byte runs:
//   K = kbase + lg*8 + [0..7]   and   K = kbase + 16 + lg*8 + [0..7]
__device__ __forceinline__ v16bf frag_a(const __bf16* rowp, int lg, int kbase) {
  v8bf lo = *(const v8bf*)(rowp + kbase + lg * 8);
  v8bf hi = *(const v8bf*)(rowp + kbase + 16 + lg * 8);
  v16bf a;
#pragma unroll
  for (int i = 0; i < 8; ++i) { a[i] = lo[i]; a[8 + i] = hi[i]; }
  return a;
}

// B-fragment (32x16 bf16, K x N) read from a [n][k] (transposed) layout:
// lane holds col n = lane&15; K = kbase + lg*8 + [0..7] and kbase + 16 + lg*8 + [0..7].
__device__ __forceinline__ v16bf frag_b(const __bf16* rowp, int lg, int kbase) {
  v8bf lo = *(const v8bf*)(rowp + kbase + lg * 8);
  v8bf hi = *(const v8bf*)(rowp + kbase + 16 + lg * 8);
  v16bf b;
#pragma unroll
  for (int i = 0; i < 8; ++i) { b[i] = lo[i]; b[8 + i] = hi[i]; }
  return b;
}

// ---------------------------------------------------------------------------
// fp32 -> bf16 conversion (vectorized x4)
// ---------------------------------------------------------------------------
__global__ void cvt_f32_bf16(const float* __restrict__ src,
                             __bf16* __restrict__ dst, int n4) {
  int i = blockIdx.x * blockDim.x + threadIdx.x;
  if (i >= n4) return;
  float4 x = ((const float4*)src)[i];
  v4bf y;
  y[0] = (__bf16)x.x; y[1] = (__bf16)x.y; y[2] = (__bf16)x.z; y[3] = (__bf16)x.w;
  *(v4bf*)(dst + 4 * i) = y;
}

// ---------------------------------------------------------------------------
// bf16 WMMA GEMM: C[M,N] = A[M,K] @ B[K,N] + bias[N]
// Block 256 thr = 8 waves; tile 128x128x32; wave tile 32x64 (2x4 subtiles).
// Double-buffered LDS; A tile staged by async global->LDS when available.
// ---------------------------------------------------------------------------
template <bool OUT_BF16>
__global__ __launch_bounds__(256) void gemm_bf16_k(
    const __bf16* __restrict__ A, const __bf16* __restrict__ B,
    const float* __restrict__ bias, void* __restrict__ Cout,
    int M, int N, int K) {
  constexpr int BM = 128, BN = 128, BK = 32;
  constexpr int LDA = 40, LDB = 40;              // +8 pad: conflict-free b128
  __shared__ __bf16 As[2][BM * LDA];
  __shared__ __bf16 Bs[2][BN * LDB];             // transposed: [n][k]

  const int m0 = blockIdx.y * BM, n0 = blockIdx.x * BN;
  const int t = threadIdx.x;
  const int lane = t & 31, w = t >> 5;
  const int wm = w >> 1, wn = w & 1;             // 4 x 2 wave grid
  const int lg = lane >> 4, ln = lane & 15;

  const int arow = t >> 1, ahalf = t & 1;        // A: 128 rows x 2 halves
  const int bkp = (t >> 4) * 2, bnseg = t & 15;  // B: 16 k-pairs x 16 n-segs
  const __bf16* Ag = A + (size_t)(m0 + arow) * K + ahalf * 16;
  const __bf16* Bg = B + (size_t)bkp * N + n0 + bnseg * 8;

  v8f acc[2][4] = {};

  // ---- stage tile 0 ----
  {
    int4 x0 = ((const int4*)Ag)[0], x1 = ((const int4*)Ag)[1];
    int4* s = (int4*)(As[0] + arow * LDA + ahalf * 16);
    s[0] = x0; s[1] = x1;
    v8bf r0 = *(const v8bf*)Bg, r1 = *(const v8bf*)(Bg + N);
#pragma unroll
    for (int j = 0; j < 8; ++j) {
      v2bf pr; pr[0] = r0[j]; pr[1] = r1[j];
      *(v2bf*)(Bs[0] + (bnseg * 8 + j) * LDB + bkp) = pr;
    }
  }
  __syncthreads();

  const int NT = K / BK;
  for (int kt = 0; kt < NT; ++kt) {
    const int cur = kt & 1;
    const bool more = (kt + 1 < NT);
    v8bf rb0, rb1;
#if !ASYNC_LDS
    int4 xa0, xa1;
#endif
    if (more) {  // issue next-slice staging early, under this slice's WMMAs
      const __bf16* a2 = Ag + (kt + 1) * BK;
#if ASYNC_LDS
      __bf16* ldst = As[cur ^ 1] + arow * LDA + ahalf * 16;
      async_cp16(a2, ldst);
      async_cp16(a2 + 8, ldst + 8);
#else
      xa0 = ((const int4*)a2)[0]; xa1 = ((const int4*)a2)[1];
#endif
      const __bf16* b2 = Bg + (size_t)(kt + 1) * BK * N;
      rb0 = *(const v8bf*)b2; rb1 = *(const v8bf*)(b2 + N);
    }

    // ---- compute current slice ----
    v16bf af[2], bfr[4];
#pragma unroll
    for (int i = 0; i < 2; ++i)
      af[i] = frag_a(As[cur] + (wm * 32 + i * 16 + ln) * LDA, lg, 0);
#pragma unroll
    for (int j = 0; j < 4; ++j)
      bfr[j] = frag_b(Bs[cur] + (wn * 64 + j * 16 + ln) * LDB, lg, 0);
#pragma unroll
    for (int i = 0; i < 2; ++i)
#pragma unroll
      for (int j = 0; j < 4; ++j)
        acc[i][j] = wmma_bf16(af[i], bfr[j], acc[i][j]);

    if (more) {
      const int nxt = cur ^ 1;
#if !ASYNC_LDS
      int4* s = (int4*)(As[nxt] + arow * LDA + ahalf * 16);
      s[0] = xa0; s[1] = xa1;
#endif
#pragma unroll
      for (int j = 0; j < 8; ++j) {
        v2bf pr; pr[0] = rb0[j]; pr[1] = rb1[j];
        *(v2bf*)(Bs[nxt] + (bnseg * 8 + j) * LDB + bkp) = pr;
      }
#if ASYNC_LDS
      async_wait0();
#endif
      __syncthreads();
    }
  }

  // epilogue: C lane layout -> (m = r + 8*lg, n = lane&15)
#pragma unroll
  for (int i = 0; i < 2; ++i) {
#pragma unroll
    for (int j = 0; j < 4; ++j) {
      int n = n0 + wn * 64 + j * 16 + ln;
      float bv = bias[n];
#pragma unroll
      for (int r = 0; r < 8; ++r) {
        int m = m0 + wm * 32 + i * 16 + r + 8 * lg;
        float vv = acc[i][j][r] + bv;
        if (OUT_BF16) ((__bf16*)Cout)[(size_t)m * N + n] = (__bf16)vv;
        else          ((float*)Cout)[(size_t)m * N + n] = vv;
      }
    }
  }
}

// ---------------------------------------------------------------------------
// In-place RoPE on bf16 qkv (S, 3, H, 80). cos/sin halves are duplicated,
// so only index d<40 is needed.
// ---------------------------------------------------------------------------
__global__ __launch_bounds__(256) void rope_k(__bf16* __restrict__ qkv,
                                              const float* __restrict__ cosb,
                                              const float* __restrict__ sinb) {
  int idx = blockIdx.x * blockDim.x + threadIdx.x;   // S*H*40 threads
  if (idx >= S_TOK * NH * 40) return;
  int d = idx % 40;
  int h = (idx / 40) & (NH - 1);
  int s = idx / (NH * 40);
  float c = cosb[s * HD + d], sn = sinb[s * HD + d];
  __bf16* q = qkv + (size_t)s * NQKV + h * HD;
  __bf16* k = q + DIM;
  float q1 = (float)q[d], q2 = (float)q[d + 40];
  q[d]      = (__bf16)(q1 * c - q2 * sn);
  q[d + 40] = (__bf16)(q2 * c + q1 * sn);
  float k1 = (float)k[d], k2 = (float)k[d + 40];
  k[d]      = (__bf16)(k1 * c - k2 * sn);
  k[d + 40] = (__bf16)(k2 * c + k1 * sn);
}

// ---------------------------------------------------------------------------
// Flash attention, one block per (group, head, 128 q-rows). 8 waves of 32,
// each wave owns 16 q-rows. Online softmax over 16 key-chunks of 64.
//   scores = (Q @ K^T)/80 ; P = softmax ; O = P @ V
// Q is zero-padded (80->96) in LDS; K B-fragments stream straight from global
// (Q's zero padding annihilates out-of-head columns); V is transposed into
// LDS; P round-trips through LDS to become A-fragments.
// ---------------------------------------------------------------------------
__global__ __launch_bounds__(256) void attn_k(const __bf16* __restrict__ qkv,
                                              __bf16* __restrict__ out) {
  constexpr int LDQ = 104;   // 96 + 8 pad
  constexpr int CH  = 64;    // key chunk
  constexpr int LDV = 72;    // 64 + 8 pad (keys along k)
  constexpr int LDP = 72;
  __shared__ __bf16 Qs[128 * LDQ];          // 26,624 B
  __shared__ __bf16 Vs[HD * LDV];           // 11,520 B  ([d][key])
  __shared__ __bf16 Ps[8 * 16 * LDP];       // 18,432 B  (per-wave P tiles)

  const int bid = blockIdx.x;
  const int g  = bid >> 7;
  const int h  = (bid >> 3) & (NH - 1);
  const int rb = bid & 7;
  const int s0 = g * LSEQ + rb * 128;
  const int t = threadIdx.x;
  const int lane = t & 31, w = t >> 5;
  const int lg = lane >> 4, ln = lane & 15;

  const __bf16* qbase = qkv + (size_t)s0 * NQKV + h * HD;

  // stage Q (128 x 80) + zero pad cols 80..95
#pragma unroll
  for (int i = 0; i < 5; ++i) {
    int c = t + 256 * i;                      // 1280 = 128 rows * 10 segs
    int row = c / 10, seg = c % 10;
#if ASYNC_LDS
    async_cp16(qbase + (size_t)row * NQKV + seg * 8, Qs + row * LDQ + seg * 8);
#else
    int4 x = *(const int4*)(qbase + (size_t)row * NQKV + seg * 8);
    *(int4*)(Qs + row * LDQ + seg * 8) = x;
#endif
  }
  if (t < 128) {
    int4 z = {0, 0, 0, 0};
    *(int4*)(Qs + t * LDQ + 80) = z;
    *(int4*)(Qs + t * LDQ + 88) = z;
  }
#if ASYNC_LDS
  async_wait0();
#endif
  __syncthreads();

  // Q A-fragments persist in registers (3 K-steps over padded D=96)
  v16bf aq[3];
#pragma unroll
  for (int ks = 0; ks < 3; ++ks)
    aq[ks] = frag_a(Qs + (w * 16 + ln) * LDQ, lg, ks * 32);

  float mrow[8], lrow[8];
  v8f O[5] = {};
#pragma unroll
  for (int r = 0; r < 8; ++r) { mrow[r] = -1e30f; lrow[r] = 0.f; }

  const float scale = 1.0f / (float)HD;

  for (int j = 0; j < LSEQ / CH; ++j) {
    __syncthreads();
    {  // stage V chunk transposed: Vs[d][key]; prefetch next chunk
      const __bf16* vbase = qkv + (size_t)(g * LSEQ + j * CH) * NQKV + 2 * DIM + h * HD;
#pragma unroll
      for (int i = 0; i < 3; ++i) {
        int c = t + 256 * i;                  // 640 = 64 keys * 10 segs
        if (c < CH * 10) {
          int key = c / 10, seg = c % 10;
          v8bf x = *(const v8bf*)(vbase + (size_t)key * NQKV + seg * 8);
          if (j + 1 < LSEQ / CH)  // gfx1250 global_prefetch of next V chunk
            __builtin_prefetch(vbase + (size_t)(key + CH) * NQKV + seg * 8, 0, 1);
#pragma unroll
          for (int e = 0; e < 8; ++e)
            Vs[(seg * 8 + e) * LDV + key] = x[e];
        }
      }
    }
    __syncthreads();

    // scores: 16 q-rows x 64 keys per wave; batch 4 B-frag loads per K-step
    const __bf16* kbg = qkv + (size_t)(g * LSEQ + j * CH) * NQKV + DIM + h * HD;
    v8f sc[4] = {};
#pragma unroll
    for (int ks = 0; ks < 3; ++ks) {
      v16bf bk[4];
#pragma unroll
      for (int ns = 0; ns < 4; ++ns)
        bk[ns] = frag_b(kbg + (size_t)(ns * 16 + ln) * NQKV, lg, ks * 32);
#pragma unroll
      for (int ns = 0; ns < 4; ++ns)
        sc[ns] = wmma_bf16(aq[ks], bk[ns], sc[ns]);
    }

    // online softmax (row m = r + 8*lg; reduce over 16 lanes of n)
    __bf16* pw = Ps + w * 16 * LDP;
#pragma unroll
    for (int r = 0; r < 8; ++r) {
      float mx = -1e30f;
#pragma unroll
      for (int ns = 0; ns < 4; ++ns) {
        sc[ns][r] *= scale;
        mx = fmaxf(mx, sc[ns][r]);
      }
#pragma unroll
      for (int off = 1; off < 16; off <<= 1)
        mx = fmaxf(mx, __shfl_xor(mx, off, 32));
      float mnew = fmaxf(mrow[r], mx);
      float alpha = __expf(mrow[r] - mnew);
      mrow[r] = mnew;
      lrow[r] *= alpha;
#pragma unroll
      for (int d5 = 0; d5 < 5; ++d5) O[d5][r] *= alpha;
      float psum = 0.f;
#pragma unroll
      for (int ns = 0; ns < 4; ++ns) {
        float p = __expf(sc[ns][r] - mnew);
        psum += p;
        pw[(r + 8 * lg) * LDP + ns * 16 + ln] = (__bf16)p;
      }
#pragma unroll
      for (int off = 1; off < 16; off <<= 1)
        psum += __shfl_xor(psum, off, 32);
      lrow[r] += psum;
    }

    // O += P @ V   (K-dim = 64 keys -> 2 steps; N-dim = 80 -> 5 subtiles)
#pragma unroll
    for (int ks2 = 0; ks2 < 2; ++ks2) {
      v16bf ap = frag_a(pw + ln * LDP, lg, ks2 * 32);
      v16bf bv[5];
#pragma unroll
      for (int d5 = 0; d5 < 5; ++d5)
        bv[d5] = frag_b(Vs + (d5 * 16 + ln) * LDV, lg, ks2 * 32);
#pragma unroll
      for (int d5 = 0; d5 < 5; ++d5)
        O[d5] = wmma_bf16(ap, bv[d5], O[d5]);
    }
  }

  // normalize + write (S, H*80) bf16
#pragma unroll
  for (int r = 0; r < 8; ++r) {
    float inv = 1.0f / lrow[r];
    int qrow = s0 + w * 16 + r + 8 * lg;
#pragma unroll
    for (int d5 = 0; d5 < 5; ++d5)
      out[(size_t)qrow * DIM + h * HD + d5 * 16 + ln] = (__bf16)(O[d5][r] * inv);
  }
}

// ---------------------------------------------------------------------------
// launch
// ---------------------------------------------------------------------------
extern "C" void kernel_launch(void* const* d_in, const int* in_sizes, int n_in,
                              void* d_out, int out_size, void* d_ws, size_t ws_size,
                              hipStream_t stream) {
  const float* hs     = (const float*)d_in[0];
  const float* cosb   = (const float*)d_in[1];
  const float* sinb   = (const float*)d_in[2];
  const float* w_qkv  = (const float*)d_in[3];
  const float* b_qkv  = (const float*)d_in[4];
  const float* w_proj = (const float*)d_in[5];
  const float* b_proj = (const float*)d_in[6];
  // d_in[7] = cu_seqlens: groups are uniform (L=1024), handled implicitly.
  float* out = (float*)d_out;

  const size_t n_hs    = (size_t)S_TOK * DIM;        // 20,971,520
  const size_t n_wqkv  = (size_t)DIM * NQKV;         //  4,915,200
  const size_t n_wproj = (size_t)DIM * DIM;          //  1,638,400
  const size_t n_qkv   = (size_t)S_TOK * NQKV;       // 62,914,560
  __bf16* hsb    = (__bf16*)d_ws;
  __bf16* wqkvb  = hsb + n_hs;
  __bf16* wprojb = wqkvb + n_wqkv;
  __bf16* qkvb   = wprojb + n_wproj;
  __bf16* attnb  = qkvb + n_qkv;                     // total ws ~213 MB

  // 1) fp32 -> bf16 staging
  cvt_f32_bf16<<<(int)((n_hs / 4 + 255) / 256), 256, 0, stream>>>(hs, hsb, (int)(n_hs / 4));
  cvt_f32_bf16<<<(int)((n_wqkv / 4 + 255) / 256), 256, 0, stream>>>(w_qkv, wqkvb, (int)(n_wqkv / 4));
  cvt_f32_bf16<<<(int)((n_wproj / 4 + 255) / 256), 256, 0, stream>>>(w_proj, wprojb, (int)(n_wproj / 4));

  // 2) QKV projection (bf16 out)
  gemm_bf16_k<true><<<dim3(NQKV / 128, S_TOK / 128), 256, 0, stream>>>(
      hsb, wqkvb, b_qkv, qkvb, S_TOK, NQKV, DIM);

  // 3) RoPE in place on q,k
  rope_k<<<(S_TOK * NH * 40) / 256, 256, 0, stream>>>(qkvb, cosb, sinb);

  // 4) attention: one block per (group, head, 128 q-rows)
  attn_k<<<NGRP * NH * (LSEQ / 128), 256, 0, stream>>>(qkvb, attnb);

  // 5) output projection (fp32 out + bias)
  gemm_bf16_k<false><<<dim3(DIM / 128, S_TOK / 128), 256, 0, stream>>>(
      attnb, wprojb, b_proj, out, S_TOK, DIM, DIM);
}
